// REFINE_MODULE_45732811768489
// MI455X (gfx1250) — matrix operand order repaired and checked by reading, not compiled, not run on hardware
//
#include <hip/hip_runtime.h>

typedef __attribute__((ext_vector_type(16))) _Float16 v16h;
typedef __attribute__((ext_vector_type(8)))  float    v8f;

#define N_PTS   16384
#define H_DIM   128
#define KNN     10
#define C2      256                 // 2*H
#define E_EDGES (N_PTS * KNN)       // 163840
#define TM      128                 // edges per block
#define TN      64                  // out channels per block
#define TK      32                  // K step
#define KPAD    8                   // row pad (halves) -> 80B row stride (16B aligned)
#define SLOPE   0.02f
#define BN_EPS  1e-5f

// ---------------------------------------------------------------------------
// async global->LDS copy (CDNA5, ASYNCcnt-tracked). LDS address = low 32 bits
// of the generic pointer.
// ---------------------------------------------------------------------------
__device__ __forceinline__ void async_ld_b128(void* lds_ptr, const void* gptr) {
    unsigned loff = (unsigned)(unsigned long long)lds_ptr;
    asm volatile("global_load_async_to_lds_b128 %0, %1, off"
                 :: "v"(loff), "v"(gptr) : "memory");
}
__device__ __forceinline__ void wait_async0() {
    asm volatile("s_wait_asynccnt 0x0" ::: "memory");
}

// ---------------------------------------------------------------------------
// 0a. zero stats buffer
// ---------------------------------------------------------------------------
__global__ void zero_kernel(float* __restrict__ p, int n) {
    int i = blockIdx.x * blockDim.x + threadIdx.x;
    if (i < n) p[i] = 0.0f;
}
// 0b. h (f32) -> h16 (f16), 16384x128
__global__ void cvt_h_kernel(const float* __restrict__ h,
                             _Float16* __restrict__ h16) {
    int i = blockIdx.x * 256 + threadIdx.x;
    h16[i] = (_Float16)h[i];
}
// 0c. W1 (f32, [k][n]) -> wt (f16, transposed [n][k]), 256x256
__global__ void cvt_w_kernel(const float* __restrict__ W1,
                             _Float16* __restrict__ wt) {
    int t = blockIdx.x * 256 + threadIdx.x;
    int k = t >> 8, n = t & 255;
    wt[n * C2 + k] = (_Float16)W1[k * C2 + n];
}

// ---------------------------------------------------------------------------
// 1. KNN: streaming top-10, one thread per dst node, register-resident list.
// ---------------------------------------------------------------------------
__global__ __launch_bounds__(256) void knn_kernel(const float* __restrict__ x,
                                                  int* __restrict__ idx) {
    __shared__ float sx[256], sy[256], sz[256];
    const int i  = blockIdx.x * 256 + threadIdx.x;
    const float xi = x[i * 3 + 0];
    const float yi = x[i * 3 + 1];
    const float zi = x[i * 3 + 2];

    float bd[KNN];
    int   bi[KNN];
#pragma unroll
    for (int q = 0; q < KNN; ++q) { bd[q] = 3.4e38f; bi[q] = 0; }

    for (int tile = 0; tile < N_PTS; tile += 256) {
        const int t = threadIdx.x;
        sx[t] = x[(tile + t) * 3 + 0];
        sy[t] = x[(tile + t) * 3 + 1];
        sz[t] = x[(tile + t) * 3 + 2];
        __syncthreads();
        for (int s = 0; s < 256; ++s) {
            const float dx = sx[s] - xi;
            const float dy = sy[s] - yi;
            const float dz = sz[s] - zi;
            const float d2 = fmaf(dx, dx, fmaf(dy, dy, dz * dz));
            if (d2 < bd[KNN - 1]) {
                float nd = d2; int ni = tile + s;
#pragma unroll
                for (int q = 0; q < KNN; ++q) {
                    const bool lt = nd < bd[q];
                    const float td = lt ? bd[q] : nd;
                    const int   ti = lt ? bi[q] : ni;
                    bd[q] = lt ? nd : bd[q];
                    bi[q] = lt ? ni : bi[q];
                    nd = td; ni = ti;
                }
            }
        }
        __syncthreads();
    }
#pragma unroll
    for (int q = 0; q < KNN; ++q) idx[i * KNN + q] = bi[q];
}

// ---------------------------------------------------------------------------
// 2. Fused gather + GEMM1 + LeakyReLU + BN-stats.
//    128x64 block tile, K-step 32; tiles filled with async global->LDS b128;
//    each wave register-blocks 2x2 16x16 subtiles -> 4 WMMA per K-step.
// ---------------------------------------------------------------------------
__global__ __launch_bounds__(256) void gemm1_kernel(
    const _Float16* __restrict__ h16, const _Float16* __restrict__ wt,
    const float* __restrict__ b1, const int* __restrict__ idx,
    _Float16* __restrict__ zbuf, float* __restrict__ gsum,
    float* __restrict__ gsumsq) {
    __shared__ __align__(16) _Float16 As[TM][TK + KPAD];   // 128 x 40 halves
    __shared__ __align__(16) _Float16 Bs[TN][TK + KPAD];   // wt tile [n][k]
    __shared__ int   srcA[TM], dstA[TM];
    __shared__ float lsum[TN], lsumsq[TN];

    const int tid = threadIdx.x;
    const int e0  = blockIdx.x * TM;
    const int n0  = blockIdx.y * TN;

    if (tid < TM) {
        const int e = e0 + tid;
        srcA[tid] = idx[e];            // edge e = dst*K + j
        dstA[tid] = e / KNN;
    }
    if (tid < TN) { lsum[tid] = 0.0f; lsumsq[tid] = 0.0f; }
    __syncthreads();

    const int wid  = tid >> 5;
    const int lane = tid & 31;
    const int mq   = wid & 3;          // wave's M pair: subtiles 2*mq, 2*mq+1
    const int nq   = wid >> 2;         // wave's N pair: subtiles 2*nq, 2*nq+1

    // wave32 fragment layouts used here, contiguous k = 16*(lane/16) + i (16-bit A 16x32 / B 32x16)
    const int m16 = lane & 15;
    const int kbA = (lane < 16) ? 0 : 16;
    const int nB  = lane & 15;
    const int kbB = (lane < 16) ? 0 : 16;

    v8f acc00 = {}, acc01 = {}, acc10 = {}, acc11 = {};

    // per-thread async-copy assignments (uniform across K iterations)
    const int ar  = tid >> 1;          // A row 0..127
    const int as8 = (tid & 1) * 16;    // A half-offset 0 or 16
    const int bn  = tid >> 2;          // B row 0..63
    const int bs8 = (tid & 3) * 8;     // B half-offset 0,8,16,24

    for (int k0 = 0; k0 < C2; k0 += TK) {
        { // A tile: 32 halves of concat(h16[src], h16[dst]) per row, 2 b128/thread
            const int kg = k0 + as8;   // k0 multiple of 32 -> segment never crosses 128
            const int row = (kg < H_DIM) ? srcA[ar] : dstA[ar];
            const int kk  = (kg < H_DIM) ? kg : (kg - H_DIM);
            const _Float16* gp = &h16[(size_t)row * H_DIM + kk];
            async_ld_b128(&As[ar][as8],     gp);
            async_ld_b128(&As[ar][as8 + 8], gp + 8);
        }
        { // B tile: Bs[n][k] = wt[n0+n][k0+k], 1 b128/thread
            async_ld_b128(&Bs[bn][bs8], &wt[(size_t)(n0 + bn) * C2 + k0 + bs8]);
        }
        wait_async0();
        __syncthreads();

        // A fragments: half i -> K = kbA + i, row m16
        v16h a0, a1;
        {
            const int r0 = (mq * 2 + 0) * 16 + m16;
            const int r1 = (mq * 2 + 1) * 16 + m16;
#pragma unroll
            for (int g = 0; g < 2; ++g)
#pragma unroll
                for (int p = 0; p < 8; ++p) {
                    a0[g * 8 + p] = As[r0][kbA + g * 8 + p];
                    a1[g * 8 + p] = As[r1][kbA + g * 8 + p];
                }
        }
        // B fragments: half i -> K = kbB + i, column nB
        v16h b0, b1f;
        {
            const int c0 = (nq * 2 + 0) * 16 + nB;
            const int c1 = (nq * 2 + 1) * 16 + nB;
#pragma unroll
            for (int p = 0; p < 16; ++p) {
                b0[p]  = Bs[c0][kbB + p];
                b1f[p] = Bs[c1][kbB + p];
            }
        }
        acc00 = __builtin_amdgcn_wmma_f32_16x16x32_f16(false, a0, false, b0,
                                                       (short)0, acc00, false, false);
        acc01 = __builtin_amdgcn_wmma_f32_16x16x32_f16(false, a0, false, b1f,
                                                       (short)0, acc01, false, false);
        acc10 = __builtin_amdgcn_wmma_f32_16x16x32_f16(false, a1, false, b0,
                                                       (short)0, acc10, false, false);
        acc11 = __builtin_amdgcn_wmma_f32_16x16x32_f16(false, a1, false, b1f,
                                                       (short)0, acc11, false, false);
        __syncthreads();
    }

    // epilogue: bias + LeakyReLU, store f16 z, accumulate BN stats
    const int mofs = (lane < 16) ? 0 : 8;
#pragma unroll
    for (int q = 0; q < 2; ++q) {
        const int nl = (nq * 2 + q) * 16 + nB;
        const int ch = n0 + nl;
        const float bias = b1[ch];
        float s1 = 0.0f, s2 = 0.0f;
#pragma unroll
        for (int ms = 0; ms < 2; ++ms) {
            const v8f acc = (ms == 0) ? (q == 0 ? acc00 : acc01)
                                      : (q == 0 ? acc10 : acc11);
#pragma unroll
            for (int r = 0; r < 8; ++r) {
                const int e = e0 + (mq * 2 + ms) * 16 + r + mofs;
                float zv = acc[r] + bias;
                zv = (zv >= 0.0f) ? zv : SLOPE * zv;
                zbuf[(size_t)e * C2 + ch] = (_Float16)zv;
                s1 += zv;
                s2 += zv * zv;
            }
        }
        atomicAdd(&lsum[nl], s1);      // ds_add_f32
        atomicAdd(&lsumsq[nl], s2);
    }
    __syncthreads();
    if (tid < TN) {
        atomicAdd(&gsum[n0 + tid], lsum[tid]);
        atomicAdd(&gsumsq[n0 + tid], lsumsq[tid]);
    }
}

// ---------------------------------------------------------------------------
// 3. Fold BN + W2: coef[c] = gamma*rsqrt(var+eps)*W2 ; cterm = const part
// ---------------------------------------------------------------------------
__global__ __launch_bounds__(256) void finalize_kernel(
    const float* __restrict__ gsum, const float* __restrict__ gsumsq,
    const float* __restrict__ gamma, const float* __restrict__ beta,
    const float* __restrict__ W2, const float* __restrict__ b2,
    float* __restrict__ coef, float* __restrict__ cterm) {
    __shared__ float red[256];
    const int c = threadIdx.x;
    const float invE = 1.0f / (float)E_EDGES;
    const float mu  = gsum[c] * invE;
    const float var = gsumsq[c] * invE - mu * mu;
    const float rs  = rsqrtf(var + BN_EPS);
    const float g   = gamma[c] * rs;
    const float w   = W2[c];
    coef[c] = g * w;
    red[c]  = (beta[c] - mu * g) * w;
    __syncthreads();
    for (int s = 128; s > 0; s >>= 1) {
        if (c < s) red[c] += red[c + s];
        __syncthreads();
    }
    if (c == 0) cterm[0] = red[0] + b2[0];
}

// ---------------------------------------------------------------------------
// 4. Per-node scatter-mean: r[e] = dot(z,coef)+cterm ; x_new = x + mean(r*x_dis)
// ---------------------------------------------------------------------------
__global__ __launch_bounds__(256) void update_kernel(
    const float* __restrict__ x, const _Float16* __restrict__ zbuf,
    const int* __restrict__ idx, const float* __restrict__ coef,
    const float* __restrict__ cterm, float* __restrict__ out) {
    __shared__ float cf[C2];
    const int tid = threadIdx.x;
    cf[tid] = coef[tid];
    __syncthreads();

    const int i = blockIdx.x * 256 + tid;
    const float xi = x[i * 3 + 0];
    const float yi = x[i * 3 + 1];
    const float zi = x[i * 3 + 2];
    const float ct = cterm[0];

    float ax = 0.0f, ay = 0.0f, az = 0.0f;
    for (int j = 0; j < KNN; ++j) {
        const int e = i * KNN + j;
        const _Float16* zp = &zbuf[(size_t)e * C2];
        __builtin_prefetch(zp + C2, 0, 1);      // global_prefetch_b8
        float r = ct;
#pragma unroll 8
        for (int c = 0; c < C2; ++c) r += (float)zp[c] * cf[c];
        const int s = idx[e];
        ax += r * (x[s * 3 + 0] - xi);
        ay += r * (x[s * 3 + 1] - yi);
        az += r * (x[s * 3 + 2] - zi);
    }
    const float inv = 1.0f / (float)KNN;
    out[i * 3 + 0] = xi + ax * inv;
    out[i * 3 + 1] = yi + ay * inv;
    out[i * 3 + 2] = zi + az * inv;
}

// ---------------------------------------------------------------------------
extern "C" void kernel_launch(void* const* d_in, const int* in_sizes, int n_in,
                              void* d_out, int out_size, void* d_ws, size_t ws_size,
                              hipStream_t stream) {
    (void)in_sizes; (void)n_in; (void)out_size; (void)ws_size;
    const float* x     = (const float*)d_in[0];
    const float* h     = (const float*)d_in[1];
    const float* W1    = (const float*)d_in[2];
    const float* b1    = (const float*)d_in[3];
    const float* gamma = (const float*)d_in[4];
    const float* beta  = (const float*)d_in[5];
    const float* W2    = (const float*)d_in[6];
    const float* b2    = (const float*)d_in[7];
    float* out = (float*)d_out;

    char* ws = (char*)d_ws;
    int*      idx  = (int*)(ws);                                   // 640 KB
    _Float16* h16  = (_Float16*)(ws + (size_t)1  * 1024 * 1024);   // 4 MB
    _Float16* wt   = (_Float16*)(ws + (size_t)6  * 1024 * 1024);   // 128 KB
    _Float16* zbuf = (_Float16*)(ws + (size_t)8  * 1024 * 1024);   // 80 MB
    float*    gsum = (float*)   (ws + (size_t)92 * 1024 * 1024);
    float*    gsumsq = gsum + 256;
    float*    coef   = gsum + 512;
    float*    cterm  = gsum + 768;

    zero_kernel<<<2, 256, 0, stream>>>(gsum, 512);
    cvt_h_kernel<<<(N_PTS * H_DIM) / 256, 256, 0, stream>>>(h, h16);
    cvt_w_kernel<<<(C2 * C2) / 256, 256, 0, stream>>>(W1, wt);
    knn_kernel<<<N_PTS / 256, 256, 0, stream>>>(x, idx);
    dim3 gg(E_EDGES / TM, C2 / TN);   // 1280 x 4
    gemm1_kernel<<<gg, 256, 0, stream>>>(h16, wt, b1, idx, zbuf, gsum, gsumsq);
    finalize_kernel<<<1, 256, 0, stream>>>(gsum, gsumsq, gamma, beta, W2, b2,
                                           coef, cterm);
    update_kernel<<<N_PTS / 256, 256, 0, stream>>>(x, zbuf, idx, coef, cterm, out);
}
